// MoE_6751688589853
// MI455X (gfx1250) — matrix operand order repaired and checked by reading, not compiled, not run on hardware
//
#include <hip/hip_runtime.h>
#include <stdint.h>

// ---------------------------------------------------------------- types
typedef __bf16 bf16;
typedef __attribute__((ext_vector_type(8)))  __bf16 v8bf;
typedef __attribute__((ext_vector_type(16))) __bf16 v16bf;
typedef __attribute__((ext_vector_type(8)))  float  v8f;

union Frag16 { v16bf v; v8bf h[2]; };

// ---------------------------------------------------------------- shapes
constexpr int NTOK   = 4096;          // B*T
constexpr int DMODEL = 1024;
constexpr int NEXP   = 8;
constexpr int TOPK   = 2;
constexpr int HEXP   = 704;
constexpr int SHEXP  = 1408;
constexpr int MAXSLOTS = NTOK * TOPK + NEXP * 64;   // counts padded to 64

__device__ __forceinline__ bf16 f2bf(float f) {
  union { float f; uint32_t u; } c; c.f = f;
  uint32_t r = c.u + 0x7FFFu + ((c.u >> 16) & 1u);   // round-to-nearest-even
  union { uint16_t u; bf16 b; } o; o.u = (uint16_t)(r >> 16);
  return o.b;
}

__device__ __forceinline__ float silu_f(float x) {
  return x / (1.0f + __expf(-x));
}

#define V8F_ZERO {0.f,0.f,0.f,0.f,0.f,0.f,0.f,0.f}

// ---------------------------------------------------------------- CDNA5 async copy
// DRAM -> LDS, 16B per lane, tracked by ASYNCcnt (no VGPR bounce).
__device__ __forceinline__ void async_ld_b128(void* lds_ptr, const void* gptr) {
  uint32_t lds_off = (uint32_t)(uintptr_t)lds_ptr;   // low 32 bits = LDS byte offset
  uint64_t gaddr   = (uint64_t)(uintptr_t)gptr;
  asm volatile("global_load_async_to_lds_b128 %0, %1, off"
               :: "v"(lds_off), "v"(gaddr) : "memory");
}
__device__ __forceinline__ void wait_async0() {
  asm volatile("s_wait_asynccnt 0x0" ::: "memory");
}

// ---------------------------------------------------------------- tiny kernels
__global__ void init_kernel(int* cnt, int* zerobuf) {
  if (threadIdx.x < NEXP) cnt[threadIdx.x] = 0;
  if (threadIdx.x < 16)   zerobuf[threadIdx.x] = 0;   // 64B zero row for invalid gathers
}

// exclusive prefix of counts padded up to 64 (tile granularity)
__global__ void prefix_kernel(const int* __restrict__ cnt, int* __restrict__ off64) {
  if (threadIdx.x == 0) {
    int o = 0;
    for (int e = 0; e < NEXP; ++e) { off64[e] = o; o += (cnt[e] + 63) & ~63; }
    off64[NEXP] = o;
  }
}

__global__ __launch_bounds__(256)
void copy_cvt_kernel(const float* __restrict__ src, bf16* __restrict__ dst, size_t total) {
  for (size_t i = (size_t)blockIdx.x * blockDim.x + threadIdx.x; i < total;
       i += (size_t)gridDim.x * blockDim.x)
    dst[i] = f2bf(src[i]);
}

// src [Bm][R][C] fp32 -> dst [Bm][C][R] bf16
__global__ __launch_bounds__(256)
void transpose_cvt_kernel(const float* __restrict__ src, bf16* __restrict__ dst,
                          int Bm, int R, int C) {
  size_t total = (size_t)Bm * R * C;
  size_t rc = (size_t)R * C;
  for (size_t i = (size_t)blockIdx.x * blockDim.x + threadIdx.x; i < total;
       i += (size_t)gridDim.x * blockDim.x) {
    size_t b = i / rc;
    size_t rem = i - b * rc;
    size_t r = rem / (size_t)C;
    size_t c = rem - r * (size_t)C;
    dst[b * rc + c * (size_t)R + r] = f2bf(src[i]);
  }
}

// ---------------------------------------------------------------- router
// one wave (32 lanes) per token: 8 dot products over D, softmax, top-2.
__global__ __launch_bounds__(256)
void router_kernel(const float* __restrict__ x, const float* __restrict__ Wg,
                   int* __restrict__ cnt, int* __restrict__ bucketIdx,
                   int* __restrict__ aExp, int* __restrict__ aPos,
                   float* __restrict__ aW) {
  const int lane = threadIdx.x & 31;
  const int wv   = threadIdx.x >> 5;
  const int n    = blockIdx.x * 8 + wv;

  float acc[NEXP];
#pragma unroll
  for (int e = 0; e < NEXP; ++e) acc[e] = 0.f;

  const float* xr = x + (size_t)n * DMODEL;
  for (int d = lane; d < DMODEL; d += 32) {
    float xv = xr[d];
#pragma unroll
    for (int e = 0; e < NEXP; ++e) acc[e] += xv * Wg[e * DMODEL + d];
  }
#pragma unroll
  for (int e = 0; e < NEXP; ++e)
    for (int off = 16; off > 0; off >>= 1)
      acc[e] += __shfl_xor(acc[e], off, 32);

  if (lane == 0) {
    float mx = acc[0];
#pragma unroll
    for (int e = 1; e < NEXP; ++e) mx = fmaxf(mx, acc[e]);
    float g[NEXP]; float s = 0.f;
#pragma unroll
    for (int e = 0; e < NEXP; ++e) { g[e] = __expf(acc[e] - mx); s += g[e]; }
    float inv = 1.f / s;
#pragma unroll
    for (int e = 0; e < NEXP; ++e) g[e] *= inv;

    int e0 = 0; float v0 = g[0];
#pragma unroll
    for (int e = 1; e < NEXP; ++e) if (g[e] > v0) { v0 = g[e]; e0 = e; }
    int e1 = -1; float v1 = -1.f;
#pragma unroll
    for (int e = 0; e < NEXP; ++e) if (e != e0 && g[e] > v1) { v1 = g[e]; e1 = e; }

    float denom = v0 + v1 + 1e-20f;
    float w0 = v0 / denom, w1 = v1 / denom;

    int p0 = atomicAdd(&cnt[e0], 1);
    int p1 = atomicAdd(&cnt[e1], 1);
    bucketIdx[(size_t)e0 * NTOK + p0] = n;
    bucketIdx[(size_t)e1 * NTOK + p1] = n;
    aExp[2 * n + 0] = e0; aPos[2 * n + 0] = p0; aW[2 * n + 0] = w0;
    aExp[2 * n + 1] = e1; aPos[2 * n + 1] = p1; aW[2 * n + 1] = w1;
  }
}

// ---------------------------------------------------------------- WMMA helpers
__device__ __forceinline__ void load_fragA(Frag16& fa, const bf16 (*As)[40],
                                           int wm, int lane) {
  int arow = wm * 16 + (lane & 15);
  int koff = (lane >> 4) * 8;             // lanes 0-15: K 0..7/16..23; 16-31: K 8..15/24..31
  fa.h[0] = *(const v8bf*)&As[arow][koff];
  fa.h[1] = *(const v8bf*)&As[arow][16 + koff];
}

__device__ __forceinline__ void load_fragB(Frag16& fb, const bf16 (*Bs)[40],
                                           int col, int lane) {
  int kb2 = (lane >> 4) * 8;              // lanes 0-15: K 0..7/16..23; 16-31: K 8..15/24..31
  fb.h[0] = *(const v8bf*)&Bs[col][kb2];
  fb.h[1] = *(const v8bf*)&Bs[col][kb2 + 16];
}

// ---------------------------------------------------------------- SwiGLU GEMM
// C[64x64] = silu(A*B1) .* (A*B3); A rows gathered (routed) or direct (shared).
// Double-buffered DRAM->LDS staging via global_load_async_to_lds_b128.
template<bool GATHER>
__global__ __launch_bounds__(256)
void swiglu_gemm_kernel(const bf16* __restrict__ Abf,   // [NTOK][Kd]
                        const bf16* __restrict__ B1t,   // [E][Hdim][Kd]
                        const bf16* __restrict__ B3t,
                        const int*  __restrict__ bucketIdx, // [E][NTOK]
                        const int*  __restrict__ cnt,
                        const int*  __restrict__ off64,
                        const bf16* __restrict__ zrow,  // 64B of zeros
                        bf16* __restrict__ dstAct,      // [slots|NTOK][Hdim]
                        int Kd, int Hdim, int maxMTiles) {
  const int e  = blockIdx.x / maxMTiles;
  const int mt = blockIdx.x % maxMTiles;
  const int rows0 = mt * 64;
  const int Mcount = GATHER ? cnt[e] : NTOK;
  if (rows0 >= Mcount) return;
  const int col0 = blockIdx.y * 64;
  const int slotBase = GATHER ? off64[e] : 0;

  __shared__ __align__(16) bf16 As [2][64][40];
  __shared__ __align__(16) bf16 B1s[2][64][40];
  __shared__ __align__(16) bf16 B3s[2][64][40];

  const int tid  = threadIdx.x;
  const int lane = tid & 31;
  const int wv   = tid >> 5;
  const int wm   = wv & 3;     // M sub-tile 0..3
  const int wn   = wv >> 2;    // N half 0..1

  // staging: 256 threads -> 64 rows x 4 chunks of 8 halfs (16B)
  const int r = tid >> 2;
  const int q = tid & 3;
  const int gr = rows0 + r;
  const bool vrow = gr < Mcount;
  int srcRow = 0;
  if (GATHER) srcRow = vrow ? bucketIdx[(size_t)e * NTOK + gr] : 0;
  else        srcRow = gr;                     // always valid when !GATHER
  const bf16* aCur  = vrow ? (Abf + (size_t)srcRow * Kd + q * 8) : (zrow + q * 8);
  const int   aStep = vrow ? 32 : 0;
  const bf16* b1Cur = B1t + ((size_t)e * Hdim + col0 + r) * Kd + q * 8;
  const bf16* b3Cur = B3t + ((size_t)e * Hdim + col0 + r) * Kd + q * 8;

  v8f accG0 = V8F_ZERO, accG1 = V8F_ZERO, accU0 = V8F_ZERO, accU1 = V8F_ZERO;

  // prologue: stage first k-tile into buffer 0
  async_ld_b128(&As [0][r][q * 8], aCur);
  async_ld_b128(&B1s[0][r][q * 8], b1Cur);
  async_ld_b128(&B3s[0][r][q * 8], b3Cur);

  int p = 0;
  for (int kb = 0; kb < Kd; kb += 32) {
    wait_async0();            // this lane's async writes landed in LDS
    __syncthreads();          // everyone's writes visible; prev compute done

    if (kb + 32 < Kd) {       // stage next tile into the other buffer
      aCur += aStep; b1Cur += 32; b3Cur += 32;
      async_ld_b128(&As [p ^ 1][r][q * 8], aCur);
      async_ld_b128(&B1s[p ^ 1][r][q * 8], b1Cur);
      async_ld_b128(&B3s[p ^ 1][r][q * 8], b3Cur);
    }

    Frag16 fa; load_fragA(fa, As[p], wm, lane);
    const int c0 = wn * 32;
    Frag16 fb;
    load_fragB(fb, B1s[p], c0 + (lane & 15), lane);
    accG0 = __builtin_amdgcn_wmma_f32_16x16x32_bf16(false, fa.v, false, fb.v,
                                                    (short)0, accG0, false, false);
    load_fragB(fb, B1s[p], c0 + 16 + (lane & 15), lane);
    accG1 = __builtin_amdgcn_wmma_f32_16x16x32_bf16(false, fa.v, false, fb.v,
                                                    (short)0, accG1, false, false);
    load_fragB(fb, B3s[p], c0 + (lane & 15), lane);
    accU0 = __builtin_amdgcn_wmma_f32_16x16x32_bf16(false, fa.v, false, fb.v,
                                                    (short)0, accU0, false, false);
    load_fragB(fb, B3s[p], c0 + 16 + (lane & 15), lane);
    accU1 = __builtin_amdgcn_wmma_f32_16x16x32_bf16(false, fa.v, false, fb.v,
                                                    (short)0, accU1, false, false);
    p ^= 1;
  }

  // epilogue: C/D layout VGPR rr -> M = rr + (lane>>4)*8, N = lane&15
  const int mbase = rows0 + wm * 16 + (lane >> 4) * 8;
  const int nlane = lane & 15;
#pragma unroll
  for (int rr = 0; rr < 8; ++rr) {
    int m = mbase + rr;
    if (m < Mcount) {
      size_t drow = (size_t)(slotBase + m) * Hdim;
      dstAct[drow + col0 + wn * 32 + nlane]      = f2bf(silu_f(accG0[rr]) * accU0[rr]);
      dstAct[drow + col0 + wn * 32 + 16 + nlane] = f2bf(silu_f(accG1[rr]) * accU1[rr]);
    }
  }
}

// ---------------------------------------------------------------- down GEMM
template<bool ROUTED>
__global__ __launch_bounds__(256)
void down_gemm_kernel(const bf16* __restrict__ Abf,   // [slots|NTOK][Kd]
                      const bf16* __restrict__ Bt,    // [E][Dout][Kd]
                      const int*  __restrict__ cnt,
                      const int*  __restrict__ off64,
                      float* __restrict__ Out,        // [slots|NTOK][Dout]
                      int Kd, int Dout, int maxMTiles) {
  const int e  = blockIdx.x / maxMTiles;
  const int mt = blockIdx.x % maxMTiles;
  const int rows0 = mt * 64;
  const int Mcount = ROUTED ? cnt[e] : NTOK;
  if (rows0 >= Mcount) return;
  const int col0 = blockIdx.y * 64;
  const int slotBase = ROUTED ? off64[e] : 0;

  __shared__ __align__(16) bf16 As[2][64][40];
  __shared__ __align__(16) bf16 Bs[2][64][40];

  const int tid  = threadIdx.x;
  const int lane = tid & 31;
  const int wv   = tid >> 5;
  const int wm   = wv & 3;
  const int wn   = wv >> 2;

  const int r = tid >> 2;
  const int q = tid & 3;
  const int gr = rows0 + r;
  // padded slots are in-bounds (counts rounded to 64); garbage rows never stored
  const bf16* aCur = Abf + (size_t)(slotBase + gr) * Kd + q * 8;
  const bf16* bCur = Bt + ((size_t)e * Dout + col0 + r) * Kd + q * 8;

  v8f acc0 = V8F_ZERO, acc1 = V8F_ZERO;

  async_ld_b128(&As[0][r][q * 8], aCur);
  async_ld_b128(&Bs[0][r][q * 8], bCur);

  int p = 0;
  for (int kb = 0; kb < Kd; kb += 32) {
    wait_async0();
    __syncthreads();

    if (kb + 32 < Kd) {
      aCur += 32; bCur += 32;
      async_ld_b128(&As[p ^ 1][r][q * 8], aCur);
      async_ld_b128(&Bs[p ^ 1][r][q * 8], bCur);
    }

    Frag16 fa; load_fragA(fa, As[p], wm, lane);
    const int c0 = wn * 32;
    Frag16 fb;
    load_fragB(fb, Bs[p], c0 + (lane & 15), lane);
    acc0 = __builtin_amdgcn_wmma_f32_16x16x32_bf16(false, fa.v, false, fb.v,
                                                   (short)0, acc0, false, false);
    load_fragB(fb, Bs[p], c0 + 16 + (lane & 15), lane);
    acc1 = __builtin_amdgcn_wmma_f32_16x16x32_bf16(false, fa.v, false, fb.v,
                                                   (short)0, acc1, false, false);
    p ^= 1;
  }

  const int mbase = rows0 + wm * 16 + (lane >> 4) * 8;
  const int nlane = lane & 15;
#pragma unroll
  for (int rr = 0; rr < 8; ++rr) {
    int m = mbase + rr;
    if (m < Mcount) {
      size_t orow = (size_t)(slotBase + m) * Dout;
      Out[orow + col0 + wn * 32 + nlane]      = acc0[rr];
      Out[orow + col0 + wn * 32 + 16 + nlane] = acc1[rr];
    }
  }
}

// ---------------------------------------------------------------- combine
// out[n][d] (already = shared result) += w0*y[slot0][d] + w1*y[slot1][d]
__global__ __launch_bounds__(256)
void combine_kernel(const int* __restrict__ aExp, const int* __restrict__ aPos,
                    const float* __restrict__ aW, const int* __restrict__ off64,
                    const float* __restrict__ y, float* __restrict__ out) {
  size_t total = (size_t)NTOK * DMODEL;
  for (size_t i = (size_t)blockIdx.x * blockDim.x + threadIdx.x; i < total;
       i += (size_t)gridDim.x * blockDim.x) {
    int n = (int)(i >> 10);          // DMODEL == 1024
    int d = (int)(i & 1023);
    int s0 = off64[aExp[2 * n + 0]] + aPos[2 * n + 0];
    int s1 = off64[aExp[2 * n + 1]] + aPos[2 * n + 1];
    out[i] += aW[2 * n + 0] * y[(size_t)s0 * DMODEL + d]
            + aW[2 * n + 1] * y[(size_t)s1 * DMODEL + d];
  }
}

// ---------------------------------------------------------------- launch
extern "C" void kernel_launch(void* const* d_in, const int* in_sizes, int n_in,
                              void* d_out, int out_size, void* d_ws, size_t ws_size,
                              hipStream_t stream) {
  (void)in_sizes; (void)n_in; (void)out_size; (void)ws_size;

  const float* xf   = (const float*)d_in[0];
  const float* Wgf  = (const float*)d_in[1];
  const float* W1f  = (const float*)d_in[2];
  const float* W3f  = (const float*)d_in[3];
  const float* W2f  = (const float*)d_in[4];
  const float* Ws1f = (const float*)d_in[5];
  const float* Ws3f = (const float*)d_in[6];
  const float* Ws2f = (const float*)d_in[7];
  float* out = (float*)d_out;

  char* base = (char*)d_ws;
  size_t off = 0;
  auto alloc = [&](size_t bytes) -> void* {
    void* p = base + off;
    off = (off + bytes + 255) & ~(size_t)255;
    return p;
  };

  int*   cnt       = (int*)  alloc(NEXP * 4);
  int*   off64     = (int*)  alloc((NEXP + 1) * 4);
  int*   zerobuf   = (int*)  alloc(64);
  int*   bucketIdx = (int*)  alloc((size_t)NEXP * NTOK * 4);
  int*   aExp      = (int*)  alloc((size_t)NTOK * TOPK * 4);
  int*   aPos      = (int*)  alloc((size_t)NTOK * TOPK * 4);
  float* aW        = (float*)alloc((size_t)NTOK * TOPK * 4);
  bf16*  xbf       = (bf16*) alloc((size_t)NTOK * DMODEL * 2);
  bf16*  w1t       = (bf16*) alloc((size_t)NEXP * HEXP * DMODEL * 2);   // [E][H][D]
  bf16*  w3t       = (bf16*) alloc((size_t)NEXP * HEXP * DMODEL * 2);
  bf16*  w2t       = (bf16*) alloc((size_t)NEXP * DMODEL * HEXP * 2);   // [E][D][H]
  bf16*  ws1t      = (bf16*) alloc((size_t)SHEXP * DMODEL * 2);          // [SH][D]
  bf16*  ws3t      = (bf16*) alloc((size_t)SHEXP * DMODEL * 2);
  bf16*  ws2t      = (bf16*) alloc((size_t)DMODEL * SHEXP * 2);          // [D][SH]
  bf16*  act       = (bf16*) alloc((size_t)MAXSLOTS * HEXP * 2);
  bf16*  acts      = (bf16*) alloc((size_t)NTOK * SHEXP * 2);
  float* yrt       = (float*)alloc((size_t)MAXSLOTS * DMODEL * 4);

  init_kernel<<<1, 32, 0, stream>>>(cnt, zerobuf);
  router_kernel<<<NTOK / 8, 256, 0, stream>>>(xf, Wgf, cnt, bucketIdx, aExp, aPos, aW);

  copy_cvt_kernel<<<2048, 256, 0, stream>>>(xf, xbf, (size_t)NTOK * DMODEL);
  transpose_cvt_kernel<<<4096, 256, 0, stream>>>(W1f,  w1t,  NEXP, DMODEL, HEXP);
  transpose_cvt_kernel<<<4096, 256, 0, stream>>>(W3f,  w3t,  NEXP, DMODEL, HEXP);
  transpose_cvt_kernel<<<4096, 256, 0, stream>>>(W2f,  w2t,  NEXP, HEXP, DMODEL);
  transpose_cvt_kernel<<<1024, 256, 0, stream>>>(Ws1f, ws1t, 1, DMODEL, SHEXP);
  transpose_cvt_kernel<<<1024, 256, 0, stream>>>(Ws3f, ws3t, 1, DMODEL, SHEXP);
  transpose_cvt_kernel<<<1024, 256, 0, stream>>>(Ws2f, ws2t, 1, SHEXP, DMODEL);

  prefix_kernel<<<1, 1, 0, stream>>>(cnt, off64);

  // routed experts: gate/up + SwiGLU, then down proj
  swiglu_gemm_kernel<true><<<dim3(NEXP * (NTOK / 64), HEXP / 64), 256, 0, stream>>>(
      xbf, w1t, w3t, bucketIdx, cnt, off64, (const bf16*)zerobuf, act,
      DMODEL, HEXP, NTOK / 64);
  down_gemm_kernel<true><<<dim3(NEXP * (NTOK / 64), DMODEL / 64), 256, 0, stream>>>(
      act, w2t, cnt, off64, yrt, HEXP, DMODEL, NTOK / 64);

  // shared expert: writes d_out directly (initializes every element)
  swiglu_gemm_kernel<false><<<dim3(NTOK / 64, SHEXP / 64), 256, 0, stream>>>(
      xbf, ws1t, ws3t, nullptr, nullptr, nullptr, (const bf16*)zerobuf, acts,
      DMODEL, SHEXP, NTOK / 64);
  down_gemm_kernel<false><<<dim3(NTOK / 64, DMODEL / 64), 256, 0, stream>>>(
      acts, ws2t, nullptr, nullptr, out, SHEXP, DMODEL, NTOK / 64);

  // gate-weighted combine of routed contributions
  combine_kernel<<<4096, 256, 0, stream>>>(aExp, aPos, aW, off64, yrt, out);
}
